// TopKSAE_53403623359039
// MI455X (gfx1250) — matrix operand order repaired and checked by reading, not compiled, not run on hardware
//
#include <hip/hip_runtime.h>
#include <hip/hip_bf16.h>

typedef __bf16 bf16_t;
typedef __attribute__((ext_vector_type(16))) __bf16 v16bf;
typedef __attribute__((ext_vector_type(8)))  __bf16 v8bf;
typedef __attribute__((ext_vector_type(8)))  float  v8f;
typedef __attribute__((ext_vector_type(4)))  int    v4i;

#define N_ROWS   4096
#define DMODEL   768
#define DSPARSE  24576
#define TOPK     64

#define AS_GLOBAL __attribute__((address_space(1)))
#define AS_LDS    __attribute__((address_space(3)))

#if defined(__AMDGCN__) && __has_builtin(__builtin_amdgcn_global_load_async_to_lds_b128) && \
    __has_builtin(__builtin_amdgcn_s_wait_asynccnt)
#  define HAVE_ASYNC_LDS 1
#else
#  define HAVE_ASYNC_LDS 0
#endif

// ------------------------------------------------------------------
// prep kernels: f32 -> bf16 conversions (+ x - b_dec), W_dec transpose
// ------------------------------------------------------------------
__global__ __launch_bounds__(256) void sae_cvt_x(const float* __restrict__ x,
                                                 const float* __restrict__ b_dec,
                                                 bf16_t* __restrict__ out) {
  int i = blockIdx.x * 256 + threadIdx.x;
  if (i < N_ROWS * DMODEL) out[i] = (bf16_t)(x[i] - b_dec[i % DMODEL]);
}

__global__ __launch_bounds__(256) void sae_cvt_w(const float* __restrict__ w,
                                                 bf16_t* __restrict__ out, int n) {
  int i = blockIdx.x * 256 + threadIdx.x;
  if (i < n) out[i] = (bf16_t)w[i];
}

// W_dec [768 x 24576] f32  ->  WdecT [24576 x 768] bf16
__global__ __launch_bounds__(256) void sae_wdec_t(const float* __restrict__ Wdec,
                                                  bf16_t* __restrict__ WdecT) {
  __shared__ float tile[32][33];
  const int j0 = blockIdx.x * 32;           // sparse dim
  const int d0 = blockIdx.y * 32;           // model dim
  const int tx = threadIdx.x & 31;
  const int ty = threadIdx.x >> 5;          // 0..7
#pragma unroll
  for (int r = 0; r < 4; ++r) {
    int d = d0 + ty + r * 8;
    tile[ty + r * 8][tx] = Wdec[(size_t)d * DSPARSE + j0 + tx];
  }
  __syncthreads();
#pragma unroll
  for (int r = 0; r < 4; ++r) {
    int j = j0 + ty + r * 8;
    WdecT[(size_t)j * DMODEL + d0 + tx] = (bf16_t)tile[tx][ty + r * 8];
  }
}

// ------------------------------------------------------------------
// encoder GEMM: f = relu(Xb [4096x768] * Wb[24576x768]^T + b_enc)
// bf16 WMMA 16x16x32, block tile 128x128x32, 8 waves (4 M x 2 N),
// double-buffered LDS, async global->LDS staging (ASYNCcnt pipelined).
// ------------------------------------------------------------------
#define BM 128
#define BN 128
#define BK 32
#define LDT 40   // padded LDS row (bf16 elems): 80B stride -> conflict-free b128 reads
#define KITERS (DMODEL / BK)

// copy 16 bf16 (two 16B chunks) global -> LDS, async when available.
// ldst is a native LDS (AS3) pointer so no addrspacecast null-checks.
__device__ __forceinline__ void stage_chunk(const bf16_t* gsrc, AS_LDS bf16_t* ldst) {
#if HAVE_ASYNC_LDS
  __builtin_amdgcn_global_load_async_to_lds_b128(
      (AS_GLOBAL v4i*)(void*)gsrc, (AS_LDS v4i*)ldst, 0, 0);
  __builtin_amdgcn_global_load_async_to_lds_b128(
      (AS_GLOBAL v4i*)(void*)(gsrc + 8), (AS_LDS v4i*)(ldst + 8), 0, 0);
#else
  *(AS_LDS v8bf*)ldst       = *(const v8bf*)gsrc;
  *(AS_LDS v8bf*)(ldst + 8) = *(const v8bf*)(gsrc + 8);
#endif
}

__global__ __launch_bounds__(256) void sae_enc_wmma(const bf16_t* __restrict__ Xb,
                                                    const bf16_t* __restrict__ Wb,
                                                    const float*  __restrict__ b_enc,
                                                    float* __restrict__ f_out) {
  __shared__ bf16_t Als[2][BM * LDT];
  __shared__ bf16_t Bls[2][BN * LDT];
  const int tid  = threadIdx.x;
  const int lane = tid & 31;
  const int wave = tid >> 5;
  const int n0 = blockIdx.x * BN;
  const int m0 = blockIdx.y * BM;
  const int wm = (wave & 3) * 32;           // wave's M offset in tile
  const int wn = (wave >> 2) * 64;          // wave's N offset in tile
  const int half = lane >> 4;               // 0: lanes 0-15, 1: lanes 16-31
  const int lr   = lane & 15;

  v8f acc[2][4] = {};

  // staging assignment: thread -> (row, 16-elem half of the 32-wide K slab)
  const int srow = tid >> 1;
  const int shs  = (tid & 1) * 16;
  const int soff = srow * LDT + shs;
  const bf16_t* xsrc = Xb + (size_t)(m0 + srow) * DMODEL + shs;
  const bf16_t* wsrc = Wb + (size_t)(n0 + srow) * DMODEL + shs;
  // cast directly from the __shared__ expressions -> addrspacecast folds, no null checks
  AS_LDS bf16_t* adst[2] = { (AS_LDS bf16_t*)&Als[0][soff], (AS_LDS bf16_t*)&Als[1][soff] };
  AS_LDS bf16_t* bdst[2] = { (AS_LDS bf16_t*)&Bls[0][soff], (AS_LDS bf16_t*)&Bls[1][soff] };

  // prologue: stage tile 0 into buffer 0 (4 async ops / thread)
  stage_chunk(xsrc, adst[0]);
  stage_chunk(wsrc, bdst[0]);

  for (int it = 0; it < KITERS; ++it) {
    const int cur = it & 1;
    if (it + 1 < KITERS) {
      // stage next tile into the other buffer, then wait for current tile only
      const int kn = (it + 1) * BK;
      stage_chunk(xsrc + kn, adst[cur ^ 1]);
      stage_chunk(wsrc + kn, bdst[cur ^ 1]);
#if HAVE_ASYNC_LDS
      __builtin_amdgcn_s_wait_asynccnt(4);   // oldest 4 (current tile) complete
#endif
    } else {
#if HAVE_ASYNC_LDS
      __builtin_amdgcn_s_wait_asynccnt(0);
#endif
    }
    __syncthreads();

    const bf16_t* Ab = Als[cur];
    const bf16_t* Bb = Bls[cur];

    // A fragment (16x32 bf16): lanes 0-15 hold K 0..7 & 16..23 of row M=lr,
    // lanes 16-31 hold K 8..15 & 24..31 (ISA 16-bit A layout).
    v16bf afrag[2], bfrag[4];
#pragma unroll
    for (int i = 0; i < 2; ++i) {
      const bf16_t* ap = Ab + (wm + i * 16 + lr) * LDT + half * 8;
      v8bf lo = *(const v8bf*)ap;
      v8bf hi = *(const v8bf*)(ap + 16);
#pragma unroll
      for (int e = 0; e < 8; ++e) { afrag[i][e] = lo[e]; afrag[i][8 + e] = hi[e]; }
    }
    // B fragment (32x16 bf16): lanes 0-15 = col N=lr, K 0..7 & 16..23; lanes 16-31 K 8..15 & 24..31.
#pragma unroll
    for (int j = 0; j < 4; ++j) {
      const bf16_t* bp = Bb + (wn + j * 16 + lr) * LDT + half * 8;
      v8bf lo = *(const v8bf*)bp;
      v8bf hi = *(const v8bf*)(bp + 16);
#pragma unroll
      for (int e = 0; e < 8; ++e) { bfrag[j][e] = lo[e]; bfrag[j][8 + e] = hi[e]; }
    }
#pragma unroll
    for (int i = 0; i < 2; ++i)
#pragma unroll
      for (int j = 0; j < 4; ++j)
        acc[i][j] = __builtin_amdgcn_wmma_f32_16x16x32_bf16(
            false, afrag[i], false, bfrag[j], (short)0, acc[i][j], false, false);
    __syncthreads();
  }

  // epilogue: C layout -> VGPR v holds M = v + half*8, N = lr; add bias, relu, NT store
#pragma unroll
  for (int i = 0; i < 2; ++i) {
    const int gmb = m0 + wm + i * 16 + half * 8;
#pragma unroll
    for (int j = 0; j < 4; ++j) {
      const int gn = n0 + wn + j * 16 + lr;
      const float be = b_enc[gn];
#pragma unroll
      for (int v = 0; v < 8; ++v) {
        float val = acc[i][j][v] + be;
        val = val > 0.f ? val : 0.f;
        __builtin_nontemporal_store(val, &f_out[(size_t)(gmb + v) * DSPARSE + gn]);
      }
    }
  }
}

// ------------------------------------------------------------------
// fused per-row top-k (exact radix select on positive-float bits),
// in-place sparsification of z, and sparse decode x_hat = z @ W_dec^T
// one 256-thread workgroup per row; row staged in 96KB dynamic LDS.
// ------------------------------------------------------------------
__global__ __launch_bounds__(256) void sae_topk_decode(float* __restrict__ z,
                                                       const float* __restrict__ b_dec,
                                                       const bf16_t* __restrict__ WdecT,
                                                       float* __restrict__ xhat) {
  extern __shared__ float smem[];
  float* row   = smem;                       // 24576 f32
  int*   hist  = (int*)(smem + DSPARSE);     // 256
  int*   ctrl  = hist + 256;                 // [0]=bin [1]=remaining [2]=ties [3]=count
  float* cvals = (float*)(ctrl + 4);         // 64
  int*   cidx  = (int*)(cvals + TOPK);       // 64

  const int tid = threadIdx.x;
  const size_t base = (size_t)blockIdx.x * DSPARSE;

  for (int i = tid; i < DSPARSE; i += 256) row[i] = z[base + i];
  if (tid < TOPK) { cvals[tid] = 0.f; cidx[tid] = 0; }
  if (tid == 0)   { ctrl[1] = TOPK; ctrl[2] = 0; ctrl[3] = 0; }
  __syncthreads();

  unsigned prefix = 0u, mask = 0u;
  for (int shift = 24; shift >= 0; shift -= 8) {
    hist[tid] = 0;
    __syncthreads();
    for (int i = tid; i < DSPARSE; i += 256) {
      unsigned u = __float_as_uint(row[i]);        // relu'd -> uint order == float order
      if ((u & mask) == prefix) atomicAdd(&hist[(u >> shift) & 255u], 1);
    }
    __syncthreads();
    if (tid == 0) {
      int remaining = ctrl[1], cum = 0, b = 255;
      for (; b >= 0; --b) {
        int h = hist[b];
        if (cum + h >= remaining) { remaining -= cum; break; }
        cum += h;
      }
      if (b < 0) b = 0;
      ctrl[0] = b; ctrl[1] = remaining;
    }
    __syncthreads();
    unsigned b = (unsigned)ctrl[0];
    prefix |= b << shift;
    mask   |= 255u << shift;
  }
  const unsigned Tbits = prefix;        // bits of k-th largest value
  const int rem_ties = ctrl[1];         // how many ==T entries to keep
  __syncthreads();

  for (int i = tid; i < DSPARSE; i += 256) {
    float f = row[i];
    unsigned u = __float_as_uint(f);
    bool keep = (u > Tbits);
    if (!keep && u == Tbits) {
      int t = atomicAdd(&ctrl[2], 1);
      keep = (t < rem_ties);
    }
    __builtin_nontemporal_store(keep ? f : 0.0f, &z[base + i]);
    if (keep) {
      int p = atomicAdd(&ctrl[3], 1);
      if (p < TOPK) { cvals[p] = f; cidx[p] = i; }
    }
  }
  __syncthreads();

  // sparse decode: x_hat[d] = b_dec[d] + sum_i cvals[i] * WdecT[cidx[i]][d]
  for (int d = tid; d < DMODEL; d += 256) {
    float acc = b_dec[d];
#pragma unroll 8
    for (int i = 0; i < TOPK; ++i)
      acc = fmaf(cvals[i], (float)WdecT[(size_t)cidx[i] * DMODEL + d], acc);
    xhat[(size_t)blockIdx.x * DMODEL + d] = acc;
  }
}

// ------------------------------------------------------------------
extern "C" void kernel_launch(void* const* d_in, const int* in_sizes, int n_in,
                              void* d_out, int out_size, void* d_ws, size_t ws_size,
                              hipStream_t stream) {
  (void)in_sizes; (void)n_in; (void)out_size; (void)ws_size;
  const float* x     = (const float*)d_in[0];
  const float* W_enc = (const float*)d_in[1];
  const float* b_enc = (const float*)d_in[2];
  const float* W_dec = (const float*)d_in[3];
  const float* b_dec = (const float*)d_in[4];

  float* xhat = (float*)d_out;                               // [4096 x 768]
  float* z    = (float*)d_out + (size_t)N_ROWS * DMODEL;     // [4096 x 24576]

  char* ws = (char*)d_ws;
  bf16_t* Xb   = (bf16_t*)ws;                                              // 6.29 MB
  bf16_t* Wb   = (bf16_t*)(ws + (size_t)N_ROWS * DMODEL * 2);              // 37.7 MB
  bf16_t* WdT  = (bf16_t*)(ws + (size_t)N_ROWS * DMODEL * 2
                              + (size_t)DSPARSE * DMODEL * 2);             // 37.7 MB

  sae_cvt_x<<<(N_ROWS * DMODEL + 255) / 256, 256, 0, stream>>>(x, b_dec, Xb);
  sae_cvt_w<<<(DSPARSE * DMODEL + 255) / 256, 256, 0, stream>>>(W_enc, Wb, DSPARSE * DMODEL);
  sae_wdec_t<<<dim3(DSPARSE / 32, DMODEL / 32), 256, 0, stream>>>(W_dec, WdT);

  sae_enc_wmma<<<dim3(DSPARSE / BN, N_ROWS / BM), 256, 0, stream>>>(Xb, Wb, b_enc, z);

  const size_t smem = (size_t)DSPARSE * 4 + 256 * 4 + 4 * 4 + TOPK * 4 + TOPK * 4;
  sae_topk_decode<<<N_ROWS, 256, smem, stream>>>(z, b_dec, WdT, xhat);
}
